// Decoder_81905026335239
// MI455X (gfx1250) — matrix operand order repaired and checked by reading, not compiled, not run on hardware
//
#include <hip/hip_runtime.h>

// ---------------- problem dims ----------------
#define B_   64
#define H_   512
#define V_   20000
#define Lc_  128
#define Lce_ 64
#define Lkb_ 256
#define R_   32
#define E_   128
#define T_   32
#define H3_  1536
#define SOS_ 2
#define NEG_ (-1e9f)

#define KC_  128            // K-chunk staged in LDS
#define LDA_ (KC_ + 8)      // padded row stride (halves): 272B -> bank-conflict free

typedef __attribute__((ext_vector_type(16))) __bf16 v16bf;
typedef __attribute__((ext_vector_type(8)))  __bf16 v8bf;
typedef __attribute__((ext_vector_type(8)))  float  v8f;

// ---------------------------------------------------------------------------
// bf16 GEMM: C[M,N] = act( A[M,K] @ B[N,K]^T + bias[N] )
// A,B row-major contiguous in K.  M%64==0, N%32==0, K%128==0 required.
// 256 threads = 8 waves. Block owns 64 M rows; wave owns a 64x32 tile
// (4 m-tiles x 2 n-tiles). A panel double-buffered through LDS in K-chunks;
// B streamed from global (no reuse). act: 0=none 1=relu 2=tanh.
// ---------------------------------------------------------------------------
__global__ __launch_bounds__(256) void gemm_bf16_kernel(
    const __bf16* __restrict__ A, const __bf16* __restrict__ Bw,
    const float* __restrict__ bias, float* __restrict__ C,
    __bf16* __restrict__ Cb, int M, int N, int K, int act)
{
  __shared__ __bf16 Asm[2][64 * LDA_];
  int tid  = threadIdx.x;
  int lane = tid & 31;
  int wid  = tid >> 5;
  int ntp  = blockIdx.x * 8 + wid;          // 32-column tile-pair index
  bool active = (ntp * 32 < N);             // wave-uniform
  int m0   = blockIdx.y * 64;
  int half = lane >> 4;                     // 0: lanes 0-15, 1: lanes 16-31
  int l16  = lane & 15;

  // cooperative A copy: 4 threads per row, 32 halves (64B) each
  int arow = tid >> 2;                      // 0..63
  int aseg = (tid & 3) * 32;                // half offset within chunk
  const __bf16* Arow0 = A + (size_t)(m0 + arow) * K + aseg;
  int nchunks = K / KC_;

  v8bf pf[4];
#pragma unroll
  for (int i = 0; i < 4; ++i) pf[i] = *(const v8bf*)(Arow0 + i * 8);
#pragma unroll
  for (int i = 0; i < 4; ++i)
    *(v8bf*)(&Asm[0][arow * LDA_ + aseg + i * 8]) = pf[i];
  __syncthreads();

  v8f acc[4][2] = {};
  const __bf16* Brow0 = Bw + (size_t)(ntp * 32 + l16) * K;
  const __bf16* Brow1 = Bw + (size_t)(ntp * 32 + 16 + l16) * K;

  for (int c = 0; c < nchunks; ++c) {
    int kc  = c * KC_;
    int buf = c & 1;
    if (c + 1 < nchunks) {                  // prefetch next A chunk (latency hidden)
#pragma unroll
      for (int i = 0; i < 4; ++i)
        pf[i] = *(const v8bf*)(Arow0 + kc + KC_ + i * 8);
    }
    if (active) {
#pragma unroll
      for (int k0 = 0; k0 < KC_; k0 += 32) {
        // B fragment: lane(n=l16, half) holds K = kc+k0+half*16 .. +15
        v16bf b0 = *(const v16bf*)(Brow0 + kc + k0 + half * 16);
        v16bf b1 = *(const v16bf*)(Brow1 + kc + k0 + half * 16);
        int abase = k0 + half * 16;         // A: e0..15 -> K=abase..abase+15
#pragma unroll
        for (int mt = 0; mt < 4; ++mt) {
          const __bf16* as = &Asm[buf][(mt * 16 + l16) * LDA_ + abase];
          v8bf alo = *(const v8bf*)(as);
          v8bf ahi = *(const v8bf*)(as + 8);
          v16bf af;
#pragma unroll
          for (int i = 0; i < 8; ++i) { af[i] = alo[i]; af[i + 8] = ahi[i]; }
          acc[mt][0] = __builtin_amdgcn_wmma_f32_16x16x32_bf16(
              false, af, false, b0, (short)0, acc[mt][0], false, false);
          acc[mt][1] = __builtin_amdgcn_wmma_f32_16x16x32_bf16(
              false, af, false, b1, (short)0, acc[mt][1], false, false);
        }
      }
    }
    if (c + 1 < nchunks) {
#pragma unroll
      for (int i = 0; i < 4; ++i)
        *(v8bf*)(&Asm[buf ^ 1][arow * LDA_ + aseg + i * 8]) = pf[i];
    }
    __syncthreads();
  }

  if (active) {
#pragma unroll
    for (int j = 0; j < 2; ++j) {
      int n = (ntp * 2 + j) * 16 + l16;
      float bb = bias ? bias[n] : 0.f;
#pragma unroll
      for (int mt = 0; mt < 4; ++mt) {
#pragma unroll
        for (int i = 0; i < 8; ++i) {
          int m = m0 + mt * 16 + i + half * 8;  // C/D: VGPR i -> M=i (+8 hi lanes)
          float v = acc[mt][j][i] + bb;
          if (act == 1) v = fmaxf(v, 0.f);
          else if (act == 2) v = tanhf(v);
          size_t o = (size_t)m * N + n;
          if (C)  C[o]  = v;
          if (Cb) Cb[o] = (__bf16)v;
        }
      }
    }
  }
}

// ---------------------------------------------------------------------------
__global__ void f32_to_bf16_kernel(const float* __restrict__ in,
                                   __bf16* __restrict__ out, int n) {
  int i = blockIdx.x * 256 + threadIdx.x;
  if (i < n) out[i] = (__bf16)in[i];
}

// x_bf16 = bf16(embed[inp_t]);  inp_t = SOS (t==0) else target[:,t-1]
__global__ void gather_embed_kernel(const float* __restrict__ embed,
                                    const int* __restrict__ target, int t,
                                    __bf16* __restrict__ xbf) {
  int idx = blockIdx.x * 256 + threadIdx.x;
  if (idx >= B_ * H_) return;
  int b = idx / H_, i = idx % H_;
  int tok = (t == 0) ? SOS_ : target[b * T_ + t - 1];
  xbf[idx] = (__bf16)embed[(size_t)tok * H_ + i];
}

// GRU gate fusion: h' = (1-z)*n + z*h ; also emit bf16 h and cat[:,0:H]
__global__ void gru_gate_kernel(const float* __restrict__ gi,
                                const float* __restrict__ gh,
                                float* __restrict__ h, __bf16* __restrict__ hbf,
                                __bf16* __restrict__ cat_bf) {
  int idx = blockIdx.x * 256 + threadIdx.x;
  if (idx >= B_ * H_) return;
  int b = idx / H_, i = idx % H_;
  size_t o = (size_t)b * H3_ + i;
  float r = 1.f / (1.f + expf(-(gi[o] + gh[o])));
  float z = 1.f / (1.f + expf(-(gi[o + H_] + gh[o + H_])));
  float n = tanhf(gi[o + 2 * H_] + r * gh[o + 2 * H_]);
  float hn = (1.f - z) * n + z * h[idx];
  h[idx] = hn;
  hbf[idx] = (__bf16)hn;
  cat_bf[o] = (__bf16)hn;
}

// Per (b,r): row_hidden = mean of unmasked kb_entity rows with row id r
__global__ __launch_bounds__(256) void row_agg_kernel(
    const float* __restrict__ kb, const int* __restrict__ kb_row,
    const unsigned char* __restrict__ kb_mask,
    float* __restrict__ rowh, float* __restrict__ rowcnt) {
  int br = blockIdx.x, b = br / R_, r = br % R_, tid = threadIdx.x;
  __shared__ int match[Lkb_];
  __shared__ int cnt_s;
  if (tid == 0) cnt_s = 0;
  __syncthreads();
  if (tid < Lkb_) {
    int m = (!kb_mask[b * Lkb_ + tid]) && (kb_row[b * Lkb_ + tid] == r);
    match[tid] = m;
    if (m) atomicAdd(&cnt_s, 1);
  }
  __syncthreads();
  float inv = 1.f / fmaxf((float)cnt_s, 1.f);
  for (int i = tid; i < H_; i += 256) {
    float acc = 0.f;
    for (int k = 0; k < Lkb_; ++k)
      if (match[k]) acc += kb[((size_t)b * Lkb_ + k) * H_ + i];
    rowh[((size_t)b * R_ + r) * H_ + i] = acc * inv;
  }
  if (tid == 0) rowcnt[b * R_ + r] = (float)cnt_s;
}

// Context-entity attention + switch gate. One block per batch row.
__global__ __launch_bounds__(256) void ce_attn_kernel(
    const float* __restrict__ h, const float* __restrict__ proj,
    const float* __restrict__ ce, const unsigned char* __restrict__ ce_mask,
    const int* __restrict__ ce_id, const float* __restrict__ Wsw,
    const float* __restrict__ bsw, float* __restrict__ psw_out,
    float* __restrict__ ce_hidden, float* __restrict__ p_ctx) {
  int b = blockIdx.x, tid = threadIdx.x;
  __shared__ float red[256];
  __shared__ float lg[Lce_], wsm[Lce_];
  __shared__ float pe[E_];
  __shared__ float stat[2];
  float s = 0.f;
  for (int i = tid; i < H_; i += 256) s += h[b * H_ + i] * Wsw[i];
  red[tid] = s;
  __syncthreads();
  for (int st = 128; st > 0; st >>= 1) {
    if (tid < st) red[tid] += red[tid + st];
    __syncthreads();
  }
  if (tid == 0) stat[0] = 1.f / (1.f + expf(-(red[0] + bsw[0])));
  const float* qe = proj + (size_t)b * H3_;  // qe at offset 0
  if (tid < Lce_) {
    const float* kv = ce + ((size_t)b * Lce_ + tid) * H_;
    float l = 0.f;
    for (int i = 0; i < H_; ++i) l += qe[i] * kv[i];
    lg[tid] = ce_mask[b * Lce_ + tid] ? NEG_ : l;
  }
  if (tid < E_) pe[tid] = 0.f;
  __syncthreads();
  if (tid == 0) {
    float mx = -1e30f;
    for (int k = 0; k < Lce_; ++k) mx = fmaxf(mx, lg[k]);
    float sm = 0.f;
    for (int k = 0; k < Lce_; ++k) { float e = expf(lg[k] - mx); wsm[k] = e; sm += e; }
    stat[1] = 1.f / sm;
  }
  __syncthreads();
  if (tid < Lce_) {
    float w = wsm[tid] * stat[1];
    wsm[tid] = w;
    atomicAdd(&pe[ce_id[b * Lce_ + tid]], w);
  }
  __syncthreads();
  for (int i = tid; i < H_; i += 256) {
    float acc = 0.f;
    for (int k = 0; k < Lce_; ++k) acc += wsm[k] * ce[((size_t)b * Lce_ + k) * H_ + i];
    ce_hidden[b * H_ + i] = acc;
  }
  if (tid < E_) p_ctx[b * E_ + tid] = pe[tid];
  if (tid == 0) psw_out[b] = stat[0];
}

// KB attention: row softmax, segment softmax within rows, entity mix + p_entity.
__global__ __launch_bounds__(256) void kb_attn_kernel(
    const float* __restrict__ proj, const float* __restrict__ rowh,
    const float* __restrict__ rowcnt, const float* __restrict__ kb,
    const unsigned char* __restrict__ kb_mask, const int* __restrict__ kb_id,
    const int* __restrict__ kb_row, const float* __restrict__ psw_in,
    const float* __restrict__ ce_hidden, const float* __restrict__ p_ctx,
    __bf16* __restrict__ cat_bf, float* __restrict__ out_ptr) {
  int b = blockIdx.x, tid = threadIdx.x;
  __shared__ int rid[Lkb_];
  __shared__ float lkb[Lkb_];
  __shared__ float lrow[R_], wrow[R_], rmax[R_], rsum[R_];
  __shared__ float pe[E_];
  __shared__ float scale;
  const float* qk = proj + (size_t)b * H3_ + H_;  // qk at offset H
  if (tid < Lkb_)
    rid[tid] = kb_mask[b * Lkb_ + tid] ? -1 : kb_row[b * Lkb_ + tid];
  if (tid < E_) pe[tid] = 0.f;
  if (tid < R_) {
    const float* rv = rowh + ((size_t)b * R_ + tid) * H_;
    float l = 0.f;
    for (int i = 0; i < H_; ++i) l += qk[i] * rv[i];
    lrow[tid] = (rowcnt[b * R_ + tid] == 0.f) ? NEG_ : l;
  }
  {
    const float* kv = kb + ((size_t)b * Lkb_ + tid) * H_;
    float l = 0.f;
    for (int i = 0; i < H_; ++i) l += qk[i] * kv[i];
    lkb[tid] = l;
  }
  __syncthreads();
  if (tid == 0) {
    float mx = -1e30f;
    for (int r = 0; r < R_; ++r) mx = fmaxf(mx, lrow[r]);
    float s = 0.f;
    for (int r = 0; r < R_; ++r) { float e = expf(lrow[r] - mx); wrow[r] = e; s += e; }
    scale = 1.f / s;
  }
  __syncthreads();
  if (tid < R_) {
    wrow[tid] *= scale;
    float mx = -1e30f;
    for (int k = 0; k < Lkb_; ++k) if (rid[k] == tid) mx = fmaxf(mx, lkb[k]);
    float s = 0.f;
    for (int k = 0; k < Lkb_; ++k) if (rid[k] == tid) s += expf(lkb[k] - mx);
    rmax[tid] = mx;
    rsum[tid] = (s > 0.f) ? 1.f / s : 0.f;
  }
  __syncthreads();
  {
    int r = rid[tid];
    float pro = 0.f;
    if (r >= 0) pro = expf(lkb[tid] - rmax[r]) * rsum[r] * wrow[r];
    atomicAdd(&pe[kb_id[b * Lkb_ + tid]], pro);
  }
  float psw = psw_in[b];
  for (int i = tid; i < H_; i += 256) {
    float kh = 0.f;
    for (int r = 0; r < R_; ++r) kh += wrow[r] * rowh[((size_t)b * R_ + r) * H_ + i];
    float eh = (1.f - psw) * ce_hidden[b * H_ + i] + psw * kh;
    cat_bf[(size_t)b * H3_ + 2 * H_ + i] = (__bf16)eh;
  }
  __syncthreads();
  if (tid < E_)
    out_ptr[b * E_ + tid] = (1.f - psw) * p_ctx[b * E_ + tid] + psw * pe[tid];
}

// MLP attention over context outputs (mem_proj precomputed).
__global__ __launch_bounds__(256) void mlp_attn_kernel(
    const float* __restrict__ proj, const float* __restrict__ memproj,
    const float* __restrict__ co, const unsigned char* __restrict__ cmask,
    const float* __restrict__ v_attn, const float* __restrict__ b_attn,
    __bf16* __restrict__ cat_bf) {
  int b = blockIdx.x, tid = threadIdx.x;
  __shared__ float lg[Lc_], wsm[Lc_];
  __shared__ float scale;
  const float* qq = proj + (size_t)b * H3_ + 2 * H_;  // qq at offset 2H
  if (tid < Lc_) {
    const float* mp = memproj + ((size_t)b * Lc_ + tid) * H_;
    float l = 0.f;
    for (int i = 0; i < H_; ++i) l += v_attn[i] * tanhf(qq[i] + mp[i] + b_attn[i]);
    lg[tid] = cmask[b * Lc_ + tid] ? NEG_ : l;
  }
  __syncthreads();
  if (tid == 0) {
    float mx = -1e30f;
    for (int j = 0; j < Lc_; ++j) mx = fmaxf(mx, lg[j]);
    float s = 0.f;
    for (int j = 0; j < Lc_; ++j) { float e = expf(lg[j] - mx); wsm[j] = e; s += e; }
    scale = 1.f / s;
  }
  __syncthreads();
  for (int i = tid; i < H_; i += 256) {
    float acc = 0.f;
    for (int j = 0; j < Lc_; ++j) acc += wsm[j] * co[((size_t)b * Lc_ + j) * H_ + i];
    cat_bf[(size_t)b * H3_ + H_ + i] = (__bf16)(acc * scale);
  }
}

// ---------------------------------------------------------------------------
extern "C" void kernel_launch(void* const* d_in, const int* in_sizes, int n_in,
                              void* d_out, int out_size, void* d_ws, size_t ws_size,
                              hipStream_t stream) {
  const float* ctx_hidden = (const float*)d_in[0];
  const float* ctx_out    = (const float*)d_in[1];
  const float* ctx_ent    = (const float*)d_in[2];
  const float* kb_ent     = (const float*)d_in[3];
  const float* embed      = (const float*)d_in[4];
  const float* Wp    = (const float*)d_in[5];
  const float* bp    = (const float*)d_in[6];
  const float* Wih   = (const float*)d_in[7];
  const float* Whh   = (const float*)d_in[8];
  const float* bih   = (const float*)d_in[9];
  const float* bhh   = (const float*)d_in[10];
  const float* Wg_ce = (const float*)d_in[11];
  const float* Wg_kb = (const float*)d_in[12];
  const float* Wq    = (const float*)d_in[13];
  const float* Wm    = (const float*)d_in[14];
  const float* vatt  = (const float*)d_in[15];
  const float* batt  = (const float*)d_in[16];
  const float* Wsw   = (const float*)d_in[17];
  const float* bsw   = (const float*)d_in[18];
  const float* Wc    = (const float*)d_in[19];
  const float* bc    = (const float*)d_in[20];
  const float* Wv    = (const float*)d_in[21];
  const float* bv    = (const float*)d_in[22];
  const int* ce_id   = (const int*)d_in[23];
  const int* kb_id   = (const int*)d_in[24];
  const int* kb_row  = (const int*)d_in[25];
  const unsigned char* c_mask  = (const unsigned char*)d_in[26];
  const unsigned char* ce_mask = (const unsigned char*)d_in[27];
  const unsigned char* kb_mask = (const unsigned char*)d_in[28];
  const int* target  = (const int*)d_in[29];

  float* out_vocab = (float*)d_out;                               // [T,B,V]
  float* out_ptr   = out_vocab + (size_t)T_ * B_ * V_;            // [T,B,E]

  // ---- workspace carve-up (all 256B aligned) ----
  unsigned char* base = (unsigned char*)d_ws;
  size_t off = 0;
  auto alloc = [&](size_t bytes) -> void* {
    void* r = base + off;
    off = (off + bytes + 255) & ~(size_t)255;
    return r;
  };
  __bf16* wv_bf   = (__bf16*)alloc((size_t)V_ * H_ * 2);
  __bf16* wih_bf  = (__bf16*)alloc((size_t)H3_ * H_ * 2);
  __bf16* whh_bf  = (__bf16*)alloc((size_t)H3_ * H_ * 2);
  __bf16* wg3_bf  = (__bf16*)alloc((size_t)H3_ * H_ * 2);   // [Wg_ce; Wg_kb; Wq]
  __bf16* wm_bf   = (__bf16*)alloc((size_t)H_ * H_ * 2);
  __bf16* wc_bf   = (__bf16*)alloc((size_t)H_ * H3_ * 2);
  __bf16* wp_bf   = (__bf16*)alloc((size_t)H_ * H_ * 2);
  __bf16* co_bf   = (__bf16*)alloc((size_t)B_ * Lc_ * H_ * 2);
  __bf16* ch_bf   = (__bf16*)alloc((size_t)B_ * H_ * 2);
  float*  memproj = (float*)alloc((size_t)B_ * Lc_ * H_ * 4);
  float*  rowh    = (float*)alloc((size_t)B_ * R_ * H_ * 4);
  float*  rowcnt  = (float*)alloc((size_t)B_ * R_ * 4);
  float*  h       = (float*)alloc((size_t)B_ * H_ * 4);
  __bf16* hbf     = (__bf16*)alloc((size_t)B_ * H_ * 2);
  __bf16* xbf     = (__bf16*)alloc((size_t)B_ * H_ * 2);
  float*  gi      = (float*)alloc((size_t)B_ * H3_ * 4);
  float*  gh      = (float*)alloc((size_t)B_ * H3_ * 4);
  float*  proj    = (float*)alloc((size_t)B_ * H3_ * 4);    // [qe|qk|qq]
  float*  psw     = (float*)alloc((size_t)B_ * 4);
  float*  cehid   = (float*)alloc((size_t)B_ * H_ * 4);
  float*  pctx    = (float*)alloc((size_t)B_ * E_ * 4);
  __bf16* cat_bf  = (__bf16*)alloc((size_t)B_ * H3_ * 2);   // [q|vocab_attn|entity]
  float*  cout_f  = (float*)alloc((size_t)B_ * H_ * 4);
  __bf16* cout_bf = (__bf16*)alloc((size_t)B_ * H_ * 2);
  (void)ws_size; (void)in_sizes; (void)n_in; (void)out_size;

  auto conv = [&](const float* src, __bf16* dst, size_t n) {
    f32_to_bf16_kernel<<<dim3((unsigned)((n + 255) / 256)), 256, 0, stream>>>(src, dst, (int)n);
  };
  auto gemm = [&](const __bf16* A, const __bf16* Bw, const float* bias,
                  float* C, __bf16* Cb, int M, int N, int K, int act) {
    dim3 g((unsigned)((N / 32 + 7) / 8), (unsigned)(M / 64));
    gemm_bf16_kernel<<<g, 256, 0, stream>>>(A, Bw, bias, C, Cb, M, N, K, act);
  };

  // ---- precompute (time-invariant) ----
  conv(Wv, wv_bf, (size_t)V_ * H_);
  conv(Wih, wih_bf, (size_t)H3_ * H_);
  conv(Whh, whh_bf, (size_t)H3_ * H_);
  conv(Wg_ce, wg3_bf, (size_t)H_ * H_);
  conv(Wg_kb, wg3_bf + (size_t)H_ * H_, (size_t)H_ * H_);
  conv(Wq, wg3_bf + (size_t)2 * H_ * H_, (size_t)H_ * H_);
  conv(Wm, wm_bf, (size_t)H_ * H_);
  conv(Wc, wc_bf, (size_t)H_ * H3_);
  conv(Wp, wp_bf, (size_t)H_ * H_);
  conv(ctx_out, co_bf, (size_t)B_ * Lc_ * H_);
  conv(ctx_hidden, ch_bf, (size_t)B_ * H_);

  row_agg_kernel<<<B_ * R_, 256, 0, stream>>>(kb_ent, kb_row, kb_mask, rowh, rowcnt);

  // h0 = relu(ch @ Wp^T + bp)  (also emits bf16 h for the first gh GEMM)
  gemm(ch_bf, wp_bf, bp, h, hbf, B_, H_, H_, /*relu*/1);
  // mem_proj = ctx_out @ Wm^T   [B*Lc, H]
  gemm(co_bf, wm_bf, nullptr, memproj, nullptr, B_ * Lc_, H_, H_, 0);

  // ---- decode loop ----
  for (int t = 0; t < T_; ++t) {
    gather_embed_kernel<<<(B_ * H_ + 255) / 256, 256, 0, stream>>>(embed, target, t, xbf);
    gemm(xbf, wih_bf, bih, gi, nullptr, B_, H3_, H_, 0);         // gi = x @ Wih^T
    gemm(hbf, whh_bf, bhh, gh, nullptr, B_, H3_, H_, 0);         // gh = h @ Whh^T
    gru_gate_kernel<<<(B_ * H_ + 255) / 256, 256, 0, stream>>>(gi, gh, h, hbf, cat_bf);
    gemm(hbf, wg3_bf, nullptr, proj, nullptr, B_, H3_, H_, 0);   // [qe|qk|qq]
    ce_attn_kernel<<<B_, 256, 0, stream>>>(h, proj, ctx_ent, ce_mask, ce_id,
                                           Wsw, bsw, psw, cehid, pctx);
    kb_attn_kernel<<<B_, 256, 0, stream>>>(proj, rowh, rowcnt, kb_ent, kb_mask,
                                           kb_id, kb_row, psw, cehid, pctx,
                                           cat_bf, out_ptr + (size_t)t * B_ * E_);
    mlp_attn_kernel<<<B_, 256, 0, stream>>>(proj, memproj, ctx_out, c_mask,
                                            vatt, batt, cat_bf);
    // concat_out = tanh(cat @ Wc^T + bc)
    gemm(cat_bf, wc_bf, bc, cout_f, cout_bf, B_, H_, H3_, /*tanh*/2);
    // p_vocab = concat_out @ Wv^T + bv  -> d_out slice [t]
    gemm(cout_bf, wv_bf, bv, out_vocab + (size_t)t * B_ * V_, nullptr,
         B_, V_, H_, 0);
  }
}
